// GNN_27625229648544
// MI455X (gfx1250) — compile-verified, matrix operand order checked by reading, not run on hardware
//
#include <hip/hip_runtime.h>
#include <hip/hip_bf16.h>

// ---------------- problem constants (match reference) ----------------
static constexpr int kNV = 100000;
static constexpr int kNH = 20000;
static constexpr int kE  = 1600000;
static constexpr int kL  = 200000;
static constexpr int kD  = 128;
static constexpr int kH1 = 128;
static constexpr int kH2 = 32;

// ---------------- WMMA vector types ----------------
typedef __attribute__((ext_vector_type(16))) __bf16 v16bf;
typedef __attribute__((ext_vector_type(8)))  __bf16 v8bf;
typedef __attribute__((ext_vector_type(8)))  float  v8f;

// ---------------------------------------------------------------
// A-matrix (16x32 bf16, MxK) lane layout per CDNA5 ISA 7.12.2:
//   lane = half*16 + m  (m = M row 0..15, half = 0/1)
//   a[0..7]  = K = k0 + half*8 + {0..7}
//   a[8..15] = K = k0 + 16 + half*8 + {0..7}
// B-matrix (32x16 bf16, KxN): lane holds column n = lane&15 of B,
// same K striping.  For out = X @ W^T, column j of B is row j of W
// (contiguous), so B loads are two 16-byte reads from W's row.
//
// Precision: fp32 value v is split as v = hi + lo with hi = bf16(v),
// lo = bf16(v - hi).  A x B accumulated as Ahi*Bhi + Ahi*Blo + Alo*Bhi
// (3 WMMAs) -> ~2^-16 relative error, near-fp32.
// ---------------------------------------------------------------

__device__ __forceinline__ void a_split_from_f32(const float* __restrict__ row,
                                                 int k0, int half, float scale,
                                                 v16bf& hi, v16bf& lo) {
  float t[16];
  *(float4*)(t + 0)  = *(const float4*)(row + k0 + half * 8);
  *(float4*)(t + 4)  = *(const float4*)(row + k0 + half * 8 + 4);
  *(float4*)(t + 8)  = *(const float4*)(row + k0 + 16 + half * 8);
  *(float4*)(t + 12) = *(const float4*)(row + k0 + 16 + half * 8 + 4);
#pragma unroll
  for (int i = 0; i < 16; ++i) {
    float f = t[i] * scale;
    __bf16 h = (__bf16)f;
    hi[i] = h;
    lo[i] = (__bf16)(f - (float)h);
  }
}

__device__ __forceinline__ v16bf tile_from_bf16(const __bf16* __restrict__ row,
                                                int k0, int half) {
  v8bf lo = *(const v8bf*)(row + k0 + half * 8);
  v8bf hi = *(const v8bf*)(row + k0 + 16 + half * 8);
  return __builtin_shufflevector(lo, hi, 0, 1, 2, 3, 4, 5, 6, 7,
                                 8, 9, 10, 11, 12, 13, 14, 15);
}

__device__ __forceinline__ v8f wmma_bf16(v16bf a, v16bf b, v8f c) {
  return __builtin_amdgcn_wmma_f32_16x16x32_bf16(false, a, false, b,
                                                 (short)0, c, false, false);
}

// ---------------- small utility kernels ----------------
__global__ void zero_kernel(float* __restrict__ p, size_t n) {
  size_t i = (size_t)blockIdx.x * blockDim.x + threadIdx.x;
  if (i < n) p[i] = 0.0f;
}

// fp32 -> (hi, lo) bf16 split
__global__ void cvt_split_kernel(const float* __restrict__ in,
                                 __bf16* __restrict__ hi,
                                 __bf16* __restrict__ lo, int n) {
  int i = blockIdx.x * blockDim.x + threadIdx.x;
  if (i < n) {
    float f = in[i];
    __bf16 h = (__bf16)f;
    hi[i] = h;
    lo[i] = (__bf16)(f - (float)h);
  }
}

// ---------------- edge scatter: wave per edge, lane = 4 floats ----------------
__global__ __launch_bounds__(256)
void scatter_kernel(const float* __restrict__ xsrc, const int* __restrict__ src,
                    const int* __restrict__ dst, float* __restrict__ tot,
                    float* __restrict__ cnt, int E) {
  int e = (int)(((size_t)blockIdx.x * blockDim.x + threadIdx.x) >> 5);
  if (e >= E) return;
  int lane = threadIdx.x & 31;
  int s = src[e];
  int d = dst[e];
  const float4 v = *(const float4*)(xsrc + (size_t)s * kD + lane * 4);
  float* t = tot + (size_t)d * kD + lane * 4;
  unsafeAtomicAdd(t + 0, v.x);   // global_atomic_add_f32
  unsafeAtomicAdd(t + 1, v.y);
  unsafeAtomicAdd(t + 2, v.z);
  unsafeAtomicAdd(t + 3, v.w);
  if (lane == 0) unsafeAtomicAdd(cnt + d, 1.0f);
}

// ---------------- node update GEMM ----------------
// out[N,128] = (relu?)( (tot/max(cnt,1)) @ Wr^T + b_rel + x @ Wq^T )
// block = 256 threads (8 waves); block tile = 16 rows x 128 cols;
// wave w owns the 16x16 tile at cols [16w, 16w+16).
// Mean normalization folded into the A-tile load (scale = 1/max(cnt,1)).
__global__ __launch_bounds__(256)
void node_gemm_kernel(const float* __restrict__ tot, const float* __restrict__ cnt,
                      const float* __restrict__ x,
                      const __bf16* __restrict__ WrHi, const __bf16* __restrict__ WrLo,
                      const float* __restrict__ brel,
                      const __bf16* __restrict__ WqHi, const __bf16* __restrict__ WqLo,
                      float* __restrict__ out, int N, int do_relu) {
  int row0 = blockIdx.x * 16;
  if (row0 >= N) return;
  int wave = threadIdx.x >> 5;
  int lane = threadIdx.x & 31;
  int half = lane >> 4;
  int mn   = lane & 15;       // A row / B-C column within tile
  int col0 = wave * 16;

  const float*  mrow    = tot + (size_t)(row0 + mn) * kD;
  const float*  xrow    = x   + (size_t)(row0 + mn) * kD;
  const __bf16* wrh_row = WrHi + (size_t)(col0 + mn) * kD;
  const __bf16* wrl_row = WrLo + (size_t)(col0 + mn) * kD;
  const __bf16* wqh_row = WqHi + (size_t)(col0 + mn) * kD;
  const __bf16* wql_row = WqLo + (size_t)(col0 + mn) * kD;

  float rinv = 1.0f / fmaxf(cnt[row0 + mn], 1.0f);

  v8f c = {};
#pragma unroll
  for (int k0 = 0; k0 < kD; k0 += 32) {
    v16bf ah, al;
    a_split_from_f32(mrow, k0, half, rinv, ah, al);
    v16bf bh = tile_from_bf16(wrh_row, k0, half);
    v16bf bl = tile_from_bf16(wrl_row, k0, half);
    c = wmma_bf16(ah, bh, c);
    c = wmma_bf16(ah, bl, c);
    c = wmma_bf16(al, bh, c);

    a_split_from_f32(xrow, k0, half, 1.0f, ah, al);
    bh = tile_from_bf16(wqh_row, k0, half);
    bl = tile_from_bf16(wql_row, k0, half);
    c = wmma_bf16(ah, bh, c);
    c = wmma_bf16(ah, bl, c);
    c = wmma_bf16(al, bh, c);
  }

  float bias = brel[col0 + mn];
#pragma unroll
  for (int r = 0; r < 8; ++r) {
    float v = c[r] + bias;
    if (do_relu) v = fmaxf(v, 0.0f);
    out[(size_t)(row0 + r + 8 * half) * kD + col0 + mn] = v;
  }
}

// ---------------- fused link decoder ----------------
// block = 256 threads handles 16 labels:
//  e = xv[lv]-xh[lh] -> LDS (hi/lo bf16); h1 = relu(e@W1^T+b1) (8 waves);
//  h2 = relu(h1@W2^T+b2) (2 waves); out = h2.Wp + bp.
__global__ __launch_bounds__(256)
void decoder_kernel(const float* __restrict__ xv, const float* __restrict__ xh,
                    const int* __restrict__ lv, const int* __restrict__ lh,
                    const __bf16* __restrict__ W1Hi, const __bf16* __restrict__ W1Lo,
                    const float* __restrict__ b1,
                    const __bf16* __restrict__ W2Hi, const __bf16* __restrict__ W2Lo,
                    const float* __restrict__ b2,
                    const float* __restrict__ Wp, const float* __restrict__ bp,
                    float* __restrict__ out, int Ltot) {
  __shared__ __align__(16) __bf16 eh_s[16][kD];
  __shared__ __align__(16) __bf16 el_s[16][kD];
  __shared__ __align__(16) __bf16 h1h_s[16][kH1];
  __shared__ __align__(16) __bf16 h1l_s[16][kH1];
  __shared__ float h2_s[16][kH2];

  int l0 = blockIdx.x * 16;
  if (l0 >= Ltot) return;
  int tid = threadIdx.x;

  // phase 1: gather + subtract + hi/lo split into LDS
  {
    int row = tid >> 4;
    int cb  = (tid & 15) * 8;
    int iv = lv[l0 + row];
    int ih = lh[l0 + row];
    const float* pv = xv + (size_t)iv * kD + cb;
    const float* ph = xh + (size_t)ih * kD + cb;
#pragma unroll
    for (int i = 0; i < 8; ++i) {
      float e = pv[i] - ph[i];
      __bf16 h = (__bf16)e;
      eh_s[row][cb + i] = h;
      el_s[row][cb + i] = (__bf16)(e - (float)h);
    }
  }
  __syncthreads();

  int wave = tid >> 5;
  int lane = tid & 31;
  int half = lane >> 4;
  int mn   = lane & 15;

  // phase 2: h1 (128 cols, 8 waves)
  {
    int col0 = wave * 16;
    const __bf16* wh_row = W1Hi + (size_t)(col0 + mn) * kD;
    const __bf16* wl_row = W1Lo + (size_t)(col0 + mn) * kD;
    v8f c = {};
#pragma unroll
    for (int k0 = 0; k0 < kD; k0 += 32) {
      v16bf ah = tile_from_bf16(&eh_s[mn][0], k0, half);   // ds_load path
      v16bf al = tile_from_bf16(&el_s[mn][0], k0, half);
      v16bf bh = tile_from_bf16(wh_row, k0, half);
      v16bf bl = tile_from_bf16(wl_row, k0, half);
      c = wmma_bf16(ah, bh, c);
      c = wmma_bf16(ah, bl, c);
      c = wmma_bf16(al, bh, c);
    }
    float bias = b1[col0 + mn];
#pragma unroll
    for (int r = 0; r < 8; ++r) {
      float v = fmaxf(c[r] + bias, 0.0f);
      __bf16 h = (__bf16)v;
      h1h_s[r + 8 * half][col0 + mn] = h;
      h1l_s[r + 8 * half][col0 + mn] = (__bf16)(v - (float)h);
    }
  }
  __syncthreads();

  // phase 3: h2 (32 cols, waves 0..1) -- wave-uniform branch, EXEC all-1s inside
  if (wave < 2) {
    int col0 = wave * 16;
    const __bf16* wh_row = W2Hi + (size_t)(col0 + mn) * kH1;
    const __bf16* wl_row = W2Lo + (size_t)(col0 + mn) * kH1;
    v8f c = {};
#pragma unroll
    for (int k0 = 0; k0 < kH1; k0 += 32) {
      v16bf ah = tile_from_bf16(&h1h_s[mn][0], k0, half);
      v16bf al = tile_from_bf16(&h1l_s[mn][0], k0, half);
      v16bf bh = tile_from_bf16(wh_row, k0, half);
      v16bf bl = tile_from_bf16(wl_row, k0, half);
      c = wmma_bf16(ah, bh, c);
      c = wmma_bf16(ah, bl, c);
      c = wmma_bf16(al, bh, c);
    }
    float bias = b2[col0 + mn];
#pragma unroll
    for (int r = 0; r < 8; ++r)
      h2_s[r + 8 * half][col0 + mn] = fmaxf(c[r] + bias, 0.0f);
  }
  __syncthreads();

  // phase 4: final projection
  if (tid < 16) {
    float acc = bp[0];
#pragma unroll
    for (int j = 0; j < kH2; ++j) acc += h2_s[tid][j] * Wp[j];
    out[l0 + tid] = acc;
  }
}

// ---------------- host orchestration ----------------
static inline int ceil_div(long long a, long long b) { return (int)((a + b - 1) / b); }

extern "C" void kernel_launch(void* const* d_in, const int* in_sizes, int n_in,
                              void* d_out, int out_size, void* d_ws, size_t ws_size,
                              hipStream_t stream) {
  (void)in_sizes; (void)n_in; (void)out_size; (void)ws_size;

  // inputs in setup_inputs() dict order
  const float* x_virus = (const float*)d_in[0];
  const float* x_host  = (const float*)d_in[1];
  const int* src_vh = (const int*)d_in[2];
  const int* dst_vh = (const int*)d_in[3];
  const int* src_hv = (const int*)d_in[4];
  const int* dst_hv = (const int*)d_in[5];
  const int* lbl_v  = (const int*)d_in[6];
  const int* lbl_h  = (const int*)d_in[7];
  const float* W_rel_vh0 = (const float*)d_in[8];
  const float* b_rel_vh0 = (const float*)d_in[9];
  const float* W_root_vh0= (const float*)d_in[10];
  const float* W_rel_hv0 = (const float*)d_in[11];
  const float* b_rel_hv0 = (const float*)d_in[12];
  const float* W_root_hv0= (const float*)d_in[13];
  const float* W_rel_vh1 = (const float*)d_in[14];
  const float* b_rel_vh1 = (const float*)d_in[15];
  const float* W_root_vh1= (const float*)d_in[16];
  const float* W_rel_hv1 = (const float*)d_in[17];
  const float* b_rel_hv1 = (const float*)d_in[18];
  const float* W_root_hv1= (const float*)d_in[19];
  const float* W1 = (const float*)d_in[20];
  const float* b1 = (const float*)d_in[21];
  const float* W2 = (const float*)d_in[22];
  const float* b2 = (const float*)d_in[23];
  const float* Wp = (const float*)d_in[24];
  const float* bp = (const float*)d_in[25];
  float* out = (float*)d_out;

  // workspace carve-up
  char* p = (char*)d_ws;
  auto alloc = [&](size_t bytes) -> void* {
    void* r = (void*)p;
    p += (bytes + 255) & ~(size_t)255;
    return r;
  };
  float* mv   = (float*)alloc((size_t)kNV * kD * 4);
  float* mh   = (float*)alloc((size_t)kNH * kD * 4);
  float* cntv = (float*)alloc((size_t)kNV * 4);
  float* cnth = (float*)alloc((size_t)kNH * 4);
  float* xv1  = (float*)alloc((size_t)kNV * kD * 4);
  float* xh1  = (float*)alloc((size_t)kNH * kD * 4);
  float* xv2  = (float*)alloc((size_t)kNV * kD * 4);
  float* xh2  = (float*)alloc((size_t)kNH * kD * 4);

  auto walloc = [&](size_t elems) -> __bf16* { return (__bf16*)alloc(elems * 2); };
  const size_t WSZ = (size_t)kD * kD;
  __bf16 *wrvh0h = walloc(WSZ), *wrvh0l = walloc(WSZ);
  __bf16 *wqvh0h = walloc(WSZ), *wqvh0l = walloc(WSZ);
  __bf16 *wrhv0h = walloc(WSZ), *wrhv0l = walloc(WSZ);
  __bf16 *wqhv0h = walloc(WSZ), *wqhv0l = walloc(WSZ);
  __bf16 *wrvh1h = walloc(WSZ), *wrvh1l = walloc(WSZ);
  __bf16 *wqvh1h = walloc(WSZ), *wqvh1l = walloc(WSZ);
  __bf16 *wrhv1h = walloc(WSZ), *wrhv1l = walloc(WSZ);
  __bf16 *wqhv1h = walloc(WSZ), *wqhv1l = walloc(WSZ);
  __bf16 *w1h = walloc((size_t)kH1 * kD),  *w1l = walloc((size_t)kH1 * kD);
  __bf16 *w2h = walloc((size_t)kH2 * kH1), *w2l = walloc((size_t)kH2 * kH1);

  const int T = 256;

  // weight conversion fp32 -> bf16 hi/lo (once per call)
  {
    const int n = kD * kD;
    cvt_split_kernel<<<ceil_div(n, T), T, 0, stream>>>(W_rel_vh0, wrvh0h, wrvh0l, n);
    cvt_split_kernel<<<ceil_div(n, T), T, 0, stream>>>(W_root_vh0, wqvh0h, wqvh0l, n);
    cvt_split_kernel<<<ceil_div(n, T), T, 0, stream>>>(W_rel_hv0, wrhv0h, wrhv0l, n);
    cvt_split_kernel<<<ceil_div(n, T), T, 0, stream>>>(W_root_hv0, wqhv0h, wqhv0l, n);
    cvt_split_kernel<<<ceil_div(n, T), T, 0, stream>>>(W_rel_vh1, wrvh1h, wrvh1l, n);
    cvt_split_kernel<<<ceil_div(n, T), T, 0, stream>>>(W_root_vh1, wqvh1h, wqvh1l, n);
    cvt_split_kernel<<<ceil_div(n, T), T, 0, stream>>>(W_rel_hv1, wrhv1h, wrhv1l, n);
    cvt_split_kernel<<<ceil_div(n, T), T, 0, stream>>>(W_root_hv1, wqhv1h, wqhv1l, n);
    cvt_split_kernel<<<ceil_div(kH1 * kD, T), T, 0, stream>>>(W1, w1h, w1l, kH1 * kD);
    cvt_split_kernel<<<ceil_div(kH2 * kH1, T), T, 0, stream>>>(W2, w2h, w2l, kH2 * kH1);
  }

  const size_t nvd = (size_t)kNV * kD, nhd = (size_t)kNH * kD;
  const int scatter_blocks = ceil_div((long long)kE * 32, T);

  // ---------------- layer 0 ----------------
  zero_kernel<<<ceil_div((long long)nhd, T), T, 0, stream>>>(mh, nhd);
  zero_kernel<<<ceil_div(kNH, T), T, 0, stream>>>(cnth, (size_t)kNH);
  scatter_kernel<<<scatter_blocks, T, 0, stream>>>(x_virus, src_vh, dst_vh, mh, cnth, kE);

  zero_kernel<<<ceil_div((long long)nvd, T), T, 0, stream>>>(mv, nvd);
  zero_kernel<<<ceil_div(kNV, T), T, 0, stream>>>(cntv, (size_t)kNV);
  scatter_kernel<<<scatter_blocks, T, 0, stream>>>(x_host, src_hv, dst_hv, mv, cntv, kE);

  node_gemm_kernel<<<kNH / 16, T, 0, stream>>>(mh, cnth, x_host,
                                               wrvh0h, wrvh0l, b_rel_vh0,
                                               wqvh0h, wqvh0l, xh1, kNH, /*relu=*/1);
  node_gemm_kernel<<<kNV / 16, T, 0, stream>>>(mv, cntv, x_virus,
                                               wrhv0h, wrhv0l, b_rel_hv0,
                                               wqhv0h, wqhv0l, xv1, kNV, /*relu=*/1);

  // ---------------- layer 1 ----------------
  zero_kernel<<<ceil_div((long long)nhd, T), T, 0, stream>>>(mh, nhd);
  zero_kernel<<<ceil_div(kNH, T), T, 0, stream>>>(cnth, (size_t)kNH);
  scatter_kernel<<<scatter_blocks, T, 0, stream>>>(xv1, src_vh, dst_vh, mh, cnth, kE);

  zero_kernel<<<ceil_div((long long)nvd, T), T, 0, stream>>>(mv, nvd);
  zero_kernel<<<ceil_div(kNV, T), T, 0, stream>>>(cntv, (size_t)kNV);
  scatter_kernel<<<scatter_blocks, T, 0, stream>>>(xh1, src_hv, dst_hv, mv, cntv, kE);

  node_gemm_kernel<<<kNH / 16, T, 0, stream>>>(mh, cnth, xh1,
                                               wrvh1h, wrvh1l, b_rel_vh1,
                                               wqvh1h, wqvh1l, xh2, kNH, /*relu=*/0);
  node_gemm_kernel<<<kNV / 16, T, 0, stream>>>(mv, cntv, xv1,
                                               wrhv1h, wrhv1l, b_rel_hv1,
                                               wqhv1h, wqhv1l, xv2, kNV, /*relu=*/0);

  // ---------------- decoder ----------------
  decoder_kernel<<<kL / 16, T, 0, stream>>>(xv2, xh2, lbl_v, lbl_h,
                                            w1h, w1l, b1, w2h, w2l, b2,
                                            Wp, bp, out, kL);
}
